// GCN_23227183137275
// MI455X (gfx1250) — matrix operand order repaired and checked by reading, not compiled, not run on hardware
//
#include <hip/hip_runtime.h>

// ---------------------------------------------------------------------------
// Types for CDNA5 WMMA (wave32): v_wmma_f32_16x16x32_bf16
// A frag: 16 bf16 / lane (8 VGPRs), B frag: 16 bf16 / lane, C/D: 8 f32 / lane
// ---------------------------------------------------------------------------
typedef unsigned short u16;
typedef unsigned int   u32;
typedef __attribute__((ext_vector_type(16))) u16    v16u;
typedef __attribute__((ext_vector_type(16))) __bf16 v16bf;
typedef __attribute__((ext_vector_type(8)))  float  v8f;

union FragU { v16u u; v16bf b; u32 d[8]; u16 e[16]; uint4 q[2]; };

__device__ __forceinline__ u16 f2bf(float f) {
  union { float f; u32 u; } x; x.f = f;
  u32 r = x.u + 0x7FFFu + ((x.u >> 16) & 1u);   // round-to-nearest-even
  return (u16)(r >> 16);
}
__device__ __forceinline__ float bf2f(u16 b) {
  union { u32 u; float f; } x; x.u = ((u32)b) << 16;
  return x.f;
}

// ---------------------------------------------------------------------------
// bf16 GEMM:  C[M,N] = A[M,K] * B[K,N]   (A,B row-major bf16, accum f32)
// Block: 256 threads = 8 waves; block tile 128x64; K-step 32.
// Wave w covers rows (w&3)*32..+31, cols (w>>2)*32..+31  -> 2x2 WMMA tiles.
// B tile is stored TRANSPOSED in LDS (Bs_t[col][k], stride 32 u16) so both A
// and B fragments are two contiguous ds_load_b128 per lane.
// HEAD=false: bf16 out (GCN hidden, pre-bias).  HEAD=true: f32 out + bias.
// ---------------------------------------------------------------------------
#define BM 128
#define BN 64
#define KT 32

template <bool HEAD>
__global__ __launch_bounds__(256)
void k_gemm_bf16(const u16* __restrict__ A, const u16* __restrict__ B,
                 int M, int N, int K,
                 u16* __restrict__ outb, float* __restrict__ outf,
                 const float* __restrict__ bias)
{
  __shared__ __align__(16) u16 As[BM * KT];    // 8 KB, row-major [row][k]
  __shared__ __align__(16) u16 Bs_t[BN * KT];  // 4 KB, transposed [col][k]

  const int tid  = threadIdx.x;
  const int lane = tid & 31;
  const int wave = tid >> 5;
  const int wm = (wave & 3) * 32;
  const int wn = (wave >> 2) * 32;
  const long long tile_m = (long long)blockIdx.x * BM;
  const int tile_n = blockIdx.y * BN;

  v8f acc[2][2] = {};

  // cooperative-load indices: A = 128x32 bf16 (32B/thread), B = 32x64 (16B/thread)
  const int ar = tid >> 1;
  const int ac = (tid & 1) * 16;
  long long agr = tile_m + ar; if (agr >= M) agr = M - 1;   // clamp; store-guarded
  const u16* agp = A + agr * (long long)K + ac;
  const int br = tid >> 3;                 // B row (k) 0..31
  const int bc = (tid & 7) * 8;            // B col base 0..56
  const u16* bgp = B + (long long)br * N + tile_n + bc;

  // per-lane fragment addresses (constant across K loop)
  const int arow0 = wm + (lane & 15);
  const int akb   = (lane >> 4) * 16;      // A: lanes0-15 K=0..15, lanes16-31 K=16..31
  const int bcol0 = wn + (lane & 15);
  const int bkb   = (lane >> 4) * 16;      // B: lanes0-15 K=0..15, lanes16-31 K=16..31

  for (int k0 = 0; k0 < K; k0 += KT) {
    uint4 av0 = *(const uint4*)(agp + k0);
    uint4 av1 = *(const uint4*)(agp + k0 + 8);
    uint4 bv  = *(const uint4*)(bgp + (long long)k0 * N);
    // prefetch next K tile into cache while this one is consumed
    __builtin_prefetch((const void*)(agp + k0 + KT), 0, 1);
    __builtin_prefetch((const void*)(bgp + (long long)(k0 + KT) * N), 0, 1);

    *(uint4*)&As[ar * KT + ac]     = av0;
    *(uint4*)&As[ar * KT + ac + 8] = av1;
    {
      const u16* be = (const u16*)&bv;
#pragma unroll
      for (int j = 0; j < 8; ++j)          // transpose 8 u16 into Bs_t
        Bs_t[(bc + j) * KT + br] = be[j];
    }
    __syncthreads();

#pragma unroll
    for (int mt = 0; mt < 2; ++mt) {
      FragU a;
      const u16* ap = &As[(arow0 + mt * 16) * KT];
      a.q[0] = *(const uint4*)(ap + akb);        // K 0..7  (+kb)
      a.q[1] = *(const uint4*)(ap + akb + 8);    // K 8..15 (+kb)
#pragma unroll
      for (int nt = 0; nt < 2; ++nt) {
        FragU b;
        const u16* bp = &Bs_t[(bcol0 + nt * 16) * KT + bkb];
        b.q[0] = *(const uint4*)(bp);            // K 0..7  (+kb)
        b.q[1] = *(const uint4*)(bp + 8);        // K 8..15 (+kb)
        acc[mt][nt] = __builtin_amdgcn_wmma_f32_16x16x32_bf16(
            false, a.b, false, b.b, (short)0, acc[mt][nt], false, false);
      }
    }
    __syncthreads();
  }

  // Epilogue. C/D layout: lane n = l&15; VGPR r -> M=r (lanes0-15) / r+8 (16-31)
  // Fast path (block fully interior in M, uniform branch): base pointer per
  // 16x16 tile + compile-time r*N immediate offsets, no per-element guards.
  const int mloc  = (lane >> 4) * 8;
  const int ncol0 = lane & 15;
  const bool full = (tile_m + BM) <= (long long)M;
#pragma unroll
  for (int mt = 0; mt < 2; ++mt) {
    const long long gr0 = tile_m + wm + mt * 16 + mloc;   // first of 8 rows
#pragma unroll
    for (int nt = 0; nt < 2; ++nt) {
      const int ncol = tile_n + wn + nt * 16 + ncol0;
      if (HEAD) {
        const float badd = bias[ncol];
        float* op = outf + gr0 * N + ncol;
        if (full) {
#pragma unroll
          for (int r = 0; r < 8; ++r) op[r * (long long)N] = acc[mt][nt][r] + badd;
        } else {
#pragma unroll
          for (int r = 0; r < 8; ++r)
            if (gr0 + r < M) op[r * (long long)N] = acc[mt][nt][r] + badd;
        }
      } else {
        u16* op = outb + gr0 * N + ncol;
        if (full) {
#pragma unroll
          for (int r = 0; r < 8; ++r) op[r * (long long)N] = f2bf(acc[mt][nt][r]);
        } else {
#pragma unroll
          for (int r = 0; r < 8; ++r)
            if (gr0 + r < M) op[r * (long long)N] = f2bf(acc[mt][nt][r]);
        }
      }
    }
  }
}

// ---------------------------------------------------------------------------
// Elementwise / graph kernels
// ---------------------------------------------------------------------------
__global__ void k_cvt_bf16(const float* __restrict__ s, u16* __restrict__ d,
                           long long n) {
  long long i = (long long)blockIdx.x * blockDim.x + threadIdx.x;
  long long st = (long long)gridDim.x * blockDim.x;
  for (; i < n; i += st) d[i] = f2bf(s[i]);
}

__global__ void k_fill_f32(float* __restrict__ p, long long n, float v) {
  long long i = (long long)blockIdx.x * blockDim.x + threadIdx.x;
  long long st = (long long)gridDim.x * blockDim.x;
  for (; i < n; i += st) p[i] = v;
}

__global__ void k_deg_accum(const int* __restrict__ dst, float* __restrict__ deg,
                            int E) {
  long long i = (long long)blockIdx.x * blockDim.x + threadIdx.x;
  long long st = (long long)gridDim.x * blockDim.x;
  for (; i < E; i += st) atomicAdd(&deg[dst[i]], 1.0f);
}

__global__ void k_rsqrt_inplace(float* __restrict__ d, int n) {
  long long i = (long long)blockIdx.x * blockDim.x + threadIdx.x;
  long long st = (long long)gridDim.x * blockDim.x;
  for (; i < n; i += st) d[i] = rsqrtf(d[i]);
}

// Edge aggregation: agg[dst] += dinv[src]*dinv[dst] * h[src]   (h bf16, agg f32)
// tpe = F/4 threads per edge (power of two); each thread handles 4 columns
// (one 8-byte bf16x4 load + 4 f32 atomics that resolve in the 192 MB L2).
__global__ void k_scatter(const u16* __restrict__ h, const int* __restrict__ eidx,
                          const float* __restrict__ dinv, float* __restrict__ agg,
                          int E, int F, int log2tpe) {
  const long long tpe = 1ll << log2tpe;
  const long long total = (long long)E << log2tpe;
  long long idx = (long long)blockIdx.x * blockDim.x + threadIdx.x;
  const long long st = (long long)gridDim.x * blockDim.x;
  for (; idx < total; idx += st) {
    const long long e = idx >> log2tpe;
    const int c = (int)(idx & (tpe - 1)) * 4;
    const int s = eidx[e];
    const int d = eidx[E + e];
    const float w = dinv[s] * dinv[d];
    const uint2 pk = *(const uint2*)(h + (long long)s * F + c);
    float* op = agg + (long long)d * F + c;
    atomicAdd(op + 0, w * bf2f((u16)(pk.x & 0xffffu)));
    atomicAdd(op + 1, w * bf2f((u16)(pk.x >> 16)));
    atomicAdd(op + 2, w * bf2f((u16)(pk.y & 0xffffu)));
    atomicAdd(op + 3, w * bf2f((u16)(pk.y >> 16)));
  }
}

// out = relu(agg + dinv[n]^2 * h[n]  (self-loop)  + bias); write bf16 for the
// next GEMM and optionally f32 (emb output) too.
__global__ void k_finalize(const float* __restrict__ agg, const u16* __restrict__ h,
                           const float* __restrict__ dinv, const float* __restrict__ bias,
                           u16* __restrict__ xnext, float* __restrict__ emb,
                           long long total, int log2F) {
  long long i = (long long)blockIdx.x * blockDim.x + threadIdx.x;
  const long long st = (long long)gridDim.x * blockDim.x;
  const int fmask = (1 << log2F) - 1;
  for (; i < total; i += st) {
    const int n = (int)(i >> log2F);
    const int f = (int)(i & fmask);
    const float di = dinv[n];
    float v = agg[i] + di * di * bf2f(h[i]) + bias[f];
    v = fmaxf(v, 0.0f);
    xnext[i] = f2bf(v);
    if (emb) emb[i] = v;
  }
}

// ---------------------------------------------------------------------------
// Launcher
// ---------------------------------------------------------------------------
static inline long long cdiv_ll(long long a, long long b) { return (a + b - 1) / b; }

extern "C" void kernel_launch(void* const* d_in, const int* in_sizes, int n_in,
                              void* d_out, int out_size, void* d_ws, size_t ws_size,
                              hipStream_t stream)
{
  const float* x   = (const float*)d_in[0];
  const int*   ei  = (const int*)d_in[1];
  const float* W1  = (const float*)d_in[2];
  const float* b1  = (const float*)d_in[3];
  const float* W2  = (const float*)d_in[4];
  const float* b2  = (const float*)d_in[5];
  const float* W3  = (const float*)d_in[6];
  const float* b3  = (const float*)d_in[7];
  const float* Wf  = (const float*)d_in[8];
  const float* bfv = (const float*)d_in[9];

  const int Nn = 50000;
  const int fin = 1024, h1 = 512, h2 = 256, h3 = 64, fo = 64;
  const int E = in_sizes[1] / 2;

  auto alignup = [](size_t v) { return (v + 255) & ~(size_t)255; };
  char* p = (char*)d_ws;
  float* dinv = (float*)p; p += alignup((size_t)Nn * 4);
  u16* xb   = (u16*)p; p += alignup((size_t)Nn * fin * 2);
  u16* w1b  = (u16*)p; p += alignup((size_t)fin * h1 * 2);
  u16* w2b  = (u16*)p; p += alignup((size_t)h1 * h2 * 2);
  u16* w3b  = (u16*)p; p += alignup((size_t)h2 * h3 * 2);
  u16* wfb  = (u16*)p; p += alignup((size_t)h3 * fo * 2);
  u16* bufA = (u16*)p; p += alignup((size_t)Nn * h1 * 2);    // h (pre-agg), bf16
  u16* bufB = (u16*)p; p += alignup((size_t)Nn * h1 * 2);    // next-layer input
  float* agg = (float*)p; p += alignup((size_t)Nn * h1 * 4); // fits in 192MB L2

  const int TB = 256;
  auto gsz = [&](long long n) {
    long long b = cdiv_ll(n, TB);
    if (b > 1048576) b = 1048576;
    return (unsigned)b;
  };

  // bf16 copies of activations + weights
  k_cvt_bf16<<<gsz((long long)Nn * fin), TB, 0, stream>>>(x, xb, (long long)Nn * fin);
  k_cvt_bf16<<<gsz((long long)fin * h1), TB, 0, stream>>>(W1, w1b, (long long)fin * h1);
  k_cvt_bf16<<<gsz((long long)h1 * h2), TB, 0, stream>>>(W2, w2b, (long long)h1 * h2);
  k_cvt_bf16<<<gsz((long long)h2 * h3), TB, 0, stream>>>(W3, w3b, (long long)h2 * h3);
  k_cvt_bf16<<<gsz((long long)h3 * fo), TB, 0, stream>>>(Wf, wfb, (long long)h3 * fo);

  // deg = 1 (self loop) + count(dst);  dinv = rsqrt(deg)
  k_fill_f32<<<gsz(Nn), TB, 0, stream>>>(dinv, Nn, 1.0f);
  k_deg_accum<<<gsz(E), TB, 0, stream>>>(ei + E, dinv, E);
  k_rsqrt_inplace<<<gsz(Nn), TB, 0, stream>>>(dinv, Nn);

  float* logits = (float*)d_out;
  float* emb    = (float*)d_out + (size_t)Nn * fo;

  // Layer 1: 1024 -> 512
  {
    dim3 g((Nn + BM - 1) / BM, h1 / BN);
    k_gemm_bf16<false><<<g, 256, 0, stream>>>(xb, w1b, Nn, h1, fin, bufA, nullptr, nullptr);
    k_fill_f32<<<gsz((long long)Nn * h1), TB, 0, stream>>>(agg, (long long)Nn * h1, 0.0f);
    k_scatter<<<gsz((long long)E * (h1 / 4)), TB, 0, stream>>>(bufA, ei, dinv, agg, E, h1, 7);
    k_finalize<<<gsz((long long)Nn * h1), TB, 0, stream>>>(agg, bufA, dinv, b1, bufB, nullptr,
                                                           (long long)Nn * h1, 9);
  }
  // Layer 2: 512 -> 256
  {
    dim3 g((Nn + BM - 1) / BM, h2 / BN);
    k_gemm_bf16<false><<<g, 256, 0, stream>>>(bufB, w2b, Nn, h2, h1, bufA, nullptr, nullptr);
    k_fill_f32<<<gsz((long long)Nn * h2), TB, 0, stream>>>(agg, (long long)Nn * h2, 0.0f);
    k_scatter<<<gsz((long long)E * (h2 / 4)), TB, 0, stream>>>(bufA, ei, dinv, agg, E, h2, 6);
    k_finalize<<<gsz((long long)Nn * h2), TB, 0, stream>>>(agg, bufA, dinv, b2, bufB, nullptr,
                                                           (long long)Nn * h2, 8);
  }
  // Layer 3: 256 -> 64  (finalize also emits f32 emb)
  {
    dim3 g((Nn + BM - 1) / BM, h3 / BN);
    k_gemm_bf16<false><<<g, 256, 0, stream>>>(bufB, w3b, Nn, h3, h2, bufA, nullptr, nullptr);
    k_fill_f32<<<gsz((long long)Nn * h3), TB, 0, stream>>>(agg, (long long)Nn * h3, 0.0f);
    k_scatter<<<gsz((long long)E * (h3 / 4)), TB, 0, stream>>>(bufA, ei, dinv, agg, E, h3, 4);
    k_finalize<<<gsz((long long)Nn * h3), TB, 0, stream>>>(agg, bufA, dinv, b3, bufB, emb,
                                                           (long long)Nn * h3, 6);
  }
  // Final head: logits = h3 @ Wf + bf  (f32 out, bias, no relu)
  {
    dim3 g((Nn + BM - 1) / BM, fo / BN);
    k_gemm_bf16<true><<<g, 256, 0, stream>>>(bufB, wfb, Nn, fo, h3, nullptr, logits, bfv);
  }
  (void)n_in; (void)out_size; (void)ws_size;
}
